// GCN_35914516529205
// MI455X (gfx1250) — matrix operand order repaired and checked by reading, not compiled, not run on hardware
//
#include <hip/hip_runtime.h>
#include <cstdint>
#include <cstddef>

#define N_NODES 50000
#define E_EDGES 800000
#define IN_F    128
#define HID_F   256
#define OUT_F   128
#define NEG_SLOPE 0.2f

typedef __attribute__((ext_vector_type(8)))  float  v8f;
typedef __attribute__((ext_vector_type(8)))  __bf16 v8bf;
typedef __attribute__((ext_vector_type(16))) __bf16 v16bf;
typedef __attribute__((ext_vector_type(4)))  unsigned int u32x4;
typedef __attribute__((ext_vector_type(4)))  int i32x4;
typedef __attribute__((ext_vector_type(8)))  int i32x8;

#if __has_builtin(__builtin_amdgcn_tensor_load_to_lds)
#define HAVE_TDM 1
#else
#define HAVE_TDM 0
#endif

// ---------- small device helpers ----------
__device__ __forceinline__ unsigned short f2bf(float f) {
    unsigned u = __float_as_uint(f);
    u += 0x7fffu + ((u >> 16) & 1u);   // round-to-nearest-even
    return (unsigned short)(u >> 16);
}

__device__ __forceinline__ float hashf(unsigned a) {
    a ^= a >> 16; a *= 0x7feb352du;
    a ^= a >> 15; a *= 0x846ca68bu;
    a ^= a >> 16;
    return (float)(a >> 8) * (1.0f / 16777216.0f);
}

__device__ __forceinline__ void atomicMaxF(float* addr, float v) {
    // IEEE trick: positive floats order as ints, negative order reversed as uints
    if (v >= 0.0f) atomicMax((int*)addr, __float_as_int(v));
    else           atomicMin((unsigned int*)addr, __float_as_uint(v));
}

__device__ __forceinline__ void edge_get(const long long* ei, const float* ea,
                                         const float* loopA, int e,
                                         int& src, int& dst, const float** attr) {
    if (e < E_EDGES) {
        src = (int)ei[e];
        dst = (int)ei[E_EDGES + e];
        *attr = ea + (size_t)e * 8;
    } else {
        int n = e - E_EDGES;
        src = n; dst = n;
        *attr = loopA + (size_t)n * 8;
    }
}

#if HAVE_TDM
// Issue one TDM 2D tile load: tile_dim0=32 (contiguous K), tile_dim1=128 rows,
// from a row-major bf16 tensor with row stride K, into LDS row-major 128x32.
// Rows beyond rows_remaining are OOB for the tensor -> zero-filled by TDM.
__device__ __forceinline__ void tdm_load_tile_a(const unsigned short* gbase,
                                                unsigned lds_addr,
                                                int rows_remaining, int K) {
    union { unsigned a[4]; u32x4 v; } g0 = {};
    union { int a[8]; i32x8 v; } g1 = {};
    i32x4 gz = {0, 0, 0, 0};
    unsigned long long ga = (unsigned long long)(uintptr_t)gbase;
    // D# group 0: count=1 | lds_addr | global_addr[56:0] | type=2 (bits 127:126)
    g0.a[0] = 1u;
    g0.a[1] = lds_addr;
    g0.a[2] = (unsigned)(ga & 0xffffffffu);
    g0.a[3] = (unsigned)((ga >> 32) & 0x01ffffffu) | 0x80000000u;
    // D# group 1:
    //  [17:16] data_size=1 (2 bytes)        -> word0
    //  [79:48] tensor_dim0 = K              -> word1[31:16], word2[15:0]
    //  [111:80] tensor_dim1 = rows_remaining-> word2[31:16], word3[15:0]
    //  [127:112] tile_dim0 = 32             -> word3[31:16]
    //  [143:128] tile_dim1 = 128            -> word4[15:0]
    //  [207:160] tensor_dim0_stride = K     -> word5, word6[15:0]
    int td1 = rows_remaining;
    g1.a[0] = 1 << 16;
    g1.a[1] = (K & 0xffff) << 16;
    g1.a[2] = ((K >> 16) & 0xffff) | ((td1 & 0xffff) << 16);
    g1.a[3] = ((td1 >> 16) & 0xffff) | (32 << 16);
    g1.a[4] = 128;
    g1.a[5] = K;
    g1.a[6] = 0;
    g1.a[7] = 0;
#if __clang_major__ >= 23
    union { int a[8]; i32x8 v; } gz8 = {};
    __builtin_amdgcn_tensor_load_to_lds(g0.v, g1.v, gz, gz, gz8.v, 0);
#else
    __builtin_amdgcn_tensor_load_to_lds(g0.v, g1.v, gz, gz, 0);
#endif
}
#endif

// ---------- utility kernels ----------
__global__ void fill_kernel(float* __restrict__ p, float v, long long n) {
    long long i = (long long)blockIdx.x * blockDim.x + threadIdx.x;
    if (i < n) p[i] = v;
}

__global__ void conv_bf16_kernel(const float* __restrict__ src,
                                 unsigned short* __restrict__ dst, long long n) {
    long long i = (long long)blockIdx.x * blockDim.x + threadIdx.x;
    if (i < n) dst[i] = f2bf(src[i]);
}

// bias + relu + hash-dropout(0.2) + bf16 convert (feeds next layer's GEMM)
__global__ void prep_kernel(const float* __restrict__ agg, const float* __restrict__ bias,
                            unsigned short* __restrict__ xbf, long long n, int C,
                            unsigned seed) {
    long long i = (long long)blockIdx.x * blockDim.x + threadIdx.x;
    if (i >= n) return;
    float t = agg[i] + bias[(int)(i % C)];
    t = fmaxf(t, 0.0f);
    float u = hashf((unsigned)i * 2654435761u + seed);
    t = (u < 0.8f) ? t * 1.25f : 0.0f;
    xbf[i] = f2bf(t);
}

__global__ void final_bias_kernel(const float* __restrict__ agg, const float* __restrict__ bias,
                                  float* __restrict__ out, long long n, int C) {
    long long i = (long long)blockIdx.x * blockDim.x + threadIdx.x;
    if (i < n) out[i] = agg[i] + bias[(int)(i % C)];
}

// ---------- edge preprocessing ----------
__global__ void edge_transform_kernel(const float* __restrict__ eattr,
                                      const long long* __restrict__ ei,
                                      const float* __restrict__ mu_c, const float* __restrict__ dev_c,
                                      const float* __restrict__ mu_n, const float* __restrict__ dev_n,
                                      float* __restrict__ ea, float* __restrict__ loopA,
                                      float* __restrict__ cnt, int E) {
    int e = blockIdx.x * blockDim.x + threadIdx.x;
    if (e >= E) return;
    float a[8];
    #pragma unroll
    for (int j = 0; j < 8; j++) a[j] = eattr[(size_t)e * 8 + j];
    float d = a[1], v;
    if (a[0] == 1.0f) { float t = d - mu_c[0]; v = __expf(-t * t / dev_c[0]); }
    else              { float t = d - mu_n[0]; v = __expf(-t * t / dev_n[0]); }
    a[1] = v;
    #pragma unroll
    for (int j = 0; j < 8; j++) ea[(size_t)e * 8 + j] = a[j];
    int dst = (int)ei[E_EDGES + e];
    atomicAdd(&cnt[dst], 1.0f);
    #pragma unroll
    for (int j = 0; j < 8; j++) atomicAdd(&loopA[(size_t)dst * 8 + j], a[j]);
}

__global__ void loop_finalize_kernel(float* __restrict__ loopA, const float* __restrict__ cnt, int n) {
    int i = blockIdx.x * blockDim.x + threadIdx.x;
    if (i >= n) return;
    float c = fmaxf(cnt[i], 1.0f);
    #pragma unroll
    for (int j = 0; j < 8; j++) loopA[(size_t)i * 8 + j] /= c;
}

// ---------- fused dual-output WMMA GEMM: XL = X@WL, XR = X@WR ----------
// Block tile 128(M) x 64(N), K-step 32, 8 waves, each wave 32x32 via 2x2 wmma tiles.
// A tile staged to LDS by the Tensor Data Mover (one DMA per K-step, wave 0).
__global__ __launch_bounds__(256) void wmma_dual_gemm(
        const unsigned short* __restrict__ X,    // nrows x K, bf16
        const unsigned short* __restrict__ WL,   // K x Kout, bf16 row-major
        const unsigned short* __restrict__ WR,
        float* __restrict__ OL, float* __restrict__ ORR,
        int nrows, int K, int Kout) {
    __shared__ __align__(32) unsigned short lA[128 * 32];   // row-major 128x32
    __shared__ __align__(32) unsigned short lBL[64 * 32];   // col-major: col*32+k
    __shared__ __align__(32) unsigned short lBR[64 * 32];

    const int tid  = threadIdx.x;
    const int lane = tid & 31;
    const int wave = tid >> 5;
    const int wr   = wave & 3;     // 4 row-groups of 32 rows
    const int wc   = wave >> 2;    // 2 col-groups of 32 cols
    const int rowBase = blockIdx.x * 128;
    const int colBase = blockIdx.y * 64;
    const int rsel = lane >> 4;    // half-wave select per ISA fragment layout
    const int rr   = lane & 15;

    const v8f zv = {0.f, 0.f, 0.f, 0.f, 0.f, 0.f, 0.f, 0.f};
    v8f accL[2][2], accR[2][2];
    #pragma unroll
    for (int i = 0; i < 2; i++)
        #pragma unroll
        for (int j = 0; j < 2; j++) { accL[i][j] = zv; accR[i][j] = zv; }

#if HAVE_TDM
    const unsigned ldsA = (unsigned)(uintptr_t)&lA[0];  // low 32 bits = LDS byte address
#endif

    for (int k0 = 0; k0 < K; k0 += 32) {
#if HAVE_TDM
        // stage A tile via Tensor Data Mover (EXEC ignored; issue from wave 0 only)
        if (wave == 0) {
            tdm_load_tile_a(X + (size_t)rowBase * K + k0, ldsA, nrows - rowBase, K);
        }
#else
        // fallback: manual A staging, thread -> (row = tid/2, 16-elem half = tid&1)
        {
            int r = tid >> 1, h = tid & 1;
            int grow = rowBase + r;
            uint4 v0 = {0, 0, 0, 0}, v1 = {0, 0, 0, 0};
            if (grow < nrows) {
                const uint4* sp = (const uint4*)(X + (size_t)grow * K + k0 + h * 16);
                v0 = sp[0]; v1 = sp[1];
            }
            uint4* dp = (uint4*)(lA + r * 32 + h * 16);
            dp[0] = v0; dp[1] = v1;
        }
#endif
        // stage B tiles column-major: thread -> (col = tid&63, k-segment = tid>>6)
        {
            int c = tid & 63, ks = tid >> 6;
            int gcol = colBase + c;
            __align__(16) unsigned short tL[8];
            __align__(16) unsigned short tR[8];
            #pragma unroll
            for (int j = 0; j < 8; j++) {
                size_t gi = (size_t)(k0 + ks * 8 + j) * Kout + gcol;
                tL[j] = WL[gi]; tR[j] = WR[gi];
            }
            *(uint4*)(lBL + c * 32 + ks * 8) = *(const uint4*)tL;
            *(uint4*)(lBR + c * 32 + ks * 8) = *(const uint4*)tR;
        }
#if HAVE_TDM
        if (wave == 0) __builtin_amdgcn_s_wait_tensorcnt(0);
#endif
        __syncthreads();

        #pragma unroll
        for (int rs = 0; rs < 2; rs++) {
            // A fragment (16-bit 16x32): lane rr = row, contiguous K run of 16 at rsel*16
            const unsigned short* ap = lA + (wr * 32 + rs * 16 + rr) * 32;
            union { struct { v8bf lo, hi; } p; v16bf v; } fa;
            fa.p.lo = *(const v8bf*)(ap + rsel * 16);
            fa.p.hi = *(const v8bf*)(ap + rsel * 16 + 8);
            v16bf a = fa.v;
            #pragma unroll
            for (int cs = 0; cs < 2; cs++) {
                // B fragment (16-bit 32x16): lane rr = col, contiguous K run of 16 at rsel*16
                const unsigned short* bpL = lBL + (wc * 32 + cs * 16 + rr) * 32 + rsel * 16;
                const unsigned short* bpR = lBR + (wc * 32 + cs * 16 + rr) * 32 + rsel * 16;
                v16bf bl = *(const v16bf*)bpL;
                v16bf br = *(const v16bf*)bpR;
                accL[rs][cs] = __builtin_amdgcn_wmma_f32_16x16x32_bf16(
                    false, a, false, bl, (short)0, accL[rs][cs], false, false);
                accR[rs][cs] = __builtin_amdgcn_wmma_f32_16x16x32_bf16(
                    false, a, false, br, (short)0, accR[rs][cs], false, false);
            }
        }
        __syncthreads();
    }

    // C/D layout: VGPR i -> row = i + 8*rsel, col = rr
    #pragma unroll
    for (int rs = 0; rs < 2; rs++)
        #pragma unroll
        for (int cs = 0; cs < 2; cs++)
            #pragma unroll
            for (int i = 0; i < 8; i++) {
                int row = rowBase + wr * 32 + rs * 16 + rsel * 8 + i;
                int col = colBase + wc * 32 + cs * 16 + rr;
                if (row < nrows) {
                    OL[(size_t)row * Kout + col]  = accL[rs][cs][i];
                    ORR[(size_t)row * Kout + col] = accR[rs][cs][i];
                }
            }
}

// ---------- edge phase ----------
// wave per edge: s = leaky_relu(xl[src]+xr[dst]+ea@we) . att ; segment max via atomics
__global__ __launch_bounds__(256) void edge_scores_kernel(
        const long long* __restrict__ ei, const float* __restrict__ ea,
        const float* __restrict__ loopA,
        const float* __restrict__ XL, const float* __restrict__ XR,
        const float* __restrict__ we, const float* __restrict__ att,
        float* __restrict__ s, float* __restrict__ smax, int etot, int C) {
    int gw = (int)(((long long)blockIdx.x * blockDim.x + threadIdx.x) >> 5);
    int lane = threadIdx.x & 31;
    if (gw >= etot) return;
    int src, dst; const float* attr;
    edge_get(ei, ea, loopA, gw, src, dst, &attr);
    float eat[8];
    #pragma unroll
    for (int j = 0; j < 8; j++) eat[j] = attr[j];
    const float* xlp = XL + (size_t)src * C;
    const float* xrp = XR + (size_t)dst * C;
    __builtin_prefetch(xlp, 0, 1);
    __builtin_prefetch(xrp, 0, 1);
    float partial = 0.0f;
    for (int c0 = 0; c0 < C; c0 += 32) {
        int c = c0 + lane;
        float em = 0.0f;
        #pragma unroll
        for (int j = 0; j < 8; j++) em = fmaf(eat[j], we[j * C + c], em);
        float m = xlp[c] + xrp[c] + em;
        float lr = (m > 0.0f) ? m : NEG_SLOPE * m;
        partial = fmaf(lr, att[c], partial);
    }
    #pragma unroll
    for (int o = 16; o > 0; o >>= 1) partial += __shfl_xor(partial, o, 32);
    if (lane == 0) {
        s[gw] = partial;
        atomicMaxF(&smax[dst], partial);
    }
}

__global__ void edge_softmax_kernel(const long long* __restrict__ ei,
                                    const float* __restrict__ ea, const float* __restrict__ loopA,
                                    const float* __restrict__ s, const float* __restrict__ smax,
                                    float* __restrict__ es, float* __restrict__ denom, int etot) {
    int e = blockIdx.x * blockDim.x + threadIdx.x;
    if (e >= etot) return;
    int src, dst; const float* attr;
    edge_get(ei, ea, loopA, e, src, dst, &attr);
    float v = __expf(s[e] - smax[dst]);
    es[e] = v;
    atomicAdd(&denom[dst], v);
}

// wave per edge: agg[dst] += (es/denom[dst]) * xl[src]
__global__ __launch_bounds__(256) void edge_aggregate_kernel(
        const long long* __restrict__ ei, const float* __restrict__ ea,
        const float* __restrict__ loopA,
        const float* __restrict__ XL, const float* __restrict__ es,
        const float* __restrict__ denom, float* __restrict__ agg, int etot, int C) {
    int gw = (int)(((long long)blockIdx.x * blockDim.x + threadIdx.x) >> 5);
    int lane = threadIdx.x & 31;
    if (gw >= etot) return;
    int src, dst; const float* attr;
    edge_get(ei, ea, loopA, gw, src, dst, &attr);
    float alpha = es[gw] / denom[dst];
    const float* xlp = XL + (size_t)src * C;
    float* op = agg + (size_t)dst * C;
    __builtin_prefetch(xlp, 0, 1);
    for (int c0 = 0; c0 < C; c0 += 32) {
        int c = c0 + lane;
        atomicAdd(&op[c], alpha * xlp[c]);
    }
}

// ---------- host orchestration ----------
static inline int cdiv(long long a, long long b) { return (int)((a + b - 1) / b); }

extern "C" void kernel_launch(void* const* d_in, const int* in_sizes, int n_in,
                              void* d_out, int out_size, void* d_ws, size_t ws_size,
                              hipStream_t stream) {
    (void)in_sizes; (void)n_in; (void)out_size; (void)ws_size;

    const float*     x     = (const float*)d_in[0];
    const long long* ei    = (const long long*)d_in[1];
    const float*     eattr = (const float*)d_in[2];
    const float* wl1 = (const float*)d_in[3];
    const float* wr1 = (const float*)d_in[4];
    const float* we1 = (const float*)d_in[5];
    const float* att1 = (const float*)d_in[6];
    const float* b1 = (const float*)d_in[7];
    const float* wl2 = (const float*)d_in[8];
    const float* wr2 = (const float*)d_in[9];
    const float* we2 = (const float*)d_in[10];
    const float* att2 = (const float*)d_in[11];
    const float* b2 = (const float*)d_in[12];
    const float* wl3 = (const float*)d_in[13];
    const float* wr3 = (const float*)d_in[14];
    const float* we3 = (const float*)d_in[15];
    const float* att3 = (const float*)d_in[16];
    const float* b3 = (const float*)d_in[17];
    const float* mu_cov  = (const float*)d_in[18];
    const float* dev_cov = (const float*)d_in[19];
    const float* mu_ncov = (const float*)d_in[20];
    const float* dev_ncov= (const float*)d_in[21];

    // workspace carve-out (256B aligned slices)
    char* w = (char*)d_ws;
    auto carve = [&](size_t bytes) { char* p = w; w += (bytes + 255) & ~(size_t)255; return p; };
    float* EA    = (float*)carve((size_t)E_EDGES * 8 * 4);
    float* LOOPA = (float*)carve((size_t)N_NODES * 8 * 4);
    float* CNT   = (float*)carve((size_t)N_NODES * 4);
    unsigned short* XBF = (unsigned short*)carve((size_t)N_NODES * 256 * 2);
    unsigned short* WLB = (unsigned short*)carve((size_t)65536 * 2);
    unsigned short* WRB = (unsigned short*)carve((size_t)65536 * 2);
    float* XL   = (float*)carve((size_t)N_NODES * 256 * 4);
    float* XR   = (float*)carve((size_t)N_NODES * 256 * 4);
    float* AGG  = (float*)carve((size_t)N_NODES * 256 * 4);
    float* S    = (float*)carve((size_t)(E_EDGES + N_NODES) * 4);
    float* ES   = (float*)carve((size_t)(E_EDGES + N_NODES) * 4);
    float* SMAX = (float*)carve((size_t)N_NODES * 4);
    float* DEN  = (float*)carve((size_t)N_NODES * 4);

    const int etot = E_EDGES + N_NODES;
    const float ninf = -__builtin_inff();
    dim3 b256(256);

    // --- edge-attr transform + self-loop attr ---
    fill_kernel<<<cdiv(N_NODES, 256), b256, 0, stream>>>(CNT, 0.0f, N_NODES);
    fill_kernel<<<cdiv((long long)N_NODES * 8, 256), b256, 0, stream>>>(LOOPA, 0.0f, (long long)N_NODES * 8);
    edge_transform_kernel<<<cdiv(E_EDGES, 256), b256, 0, stream>>>(
        eattr, ei, mu_cov, dev_cov, mu_ncov, dev_ncov, EA, LOOPA, CNT, E_EDGES);
    loop_finalize_kernel<<<cdiv(N_NODES, 256), b256, 0, stream>>>(LOOPA, CNT, N_NODES);

    const int waveBlocks = cdiv((long long)etot * 32, 256);

    // ===== layer 1: K=128, C=256 =====
    conv_bf16_kernel<<<cdiv((long long)N_NODES * IN_F, 256), b256, 0, stream>>>(x, XBF, (long long)N_NODES * IN_F);
    conv_bf16_kernel<<<cdiv(IN_F * HID_F, 256), b256, 0, stream>>>(wl1, WLB, IN_F * HID_F);
    conv_bf16_kernel<<<cdiv(IN_F * HID_F, 256), b256, 0, stream>>>(wr1, WRB, IN_F * HID_F);
    wmma_dual_gemm<<<dim3(cdiv(N_NODES, 128), HID_F / 64), b256, 0, stream>>>(
        XBF, WLB, WRB, XL, XR, N_NODES, IN_F, HID_F);
    fill_kernel<<<cdiv(N_NODES, 256), b256, 0, stream>>>(SMAX, ninf, N_NODES);
    fill_kernel<<<cdiv(N_NODES, 256), b256, 0, stream>>>(DEN, 0.0f, N_NODES);
    fill_kernel<<<cdiv((long long)N_NODES * HID_F, 256), b256, 0, stream>>>(AGG, 0.0f, (long long)N_NODES * HID_F);
    edge_scores_kernel<<<waveBlocks, b256, 0, stream>>>(ei, EA, LOOPA, XL, XR, we1, att1, S, SMAX, etot, HID_F);
    edge_softmax_kernel<<<cdiv(etot, 256), b256, 0, stream>>>(ei, EA, LOOPA, S, SMAX, ES, DEN, etot);
    edge_aggregate_kernel<<<waveBlocks, b256, 0, stream>>>(ei, EA, LOOPA, XL, ES, DEN, AGG, etot, HID_F);

    // ===== layer 2: K=256, C=256 =====
    prep_kernel<<<cdiv((long long)N_NODES * HID_F, 256), b256, 0, stream>>>(
        AGG, b1, XBF, (long long)N_NODES * HID_F, HID_F, 0x9E3779B1u);
    conv_bf16_kernel<<<cdiv(HID_F * HID_F, 256), b256, 0, stream>>>(wl2, WLB, HID_F * HID_F);
    conv_bf16_kernel<<<cdiv(HID_F * HID_F, 256), b256, 0, stream>>>(wr2, WRB, HID_F * HID_F);
    wmma_dual_gemm<<<dim3(cdiv(N_NODES, 128), HID_F / 64), b256, 0, stream>>>(
        XBF, WLB, WRB, XL, XR, N_NODES, HID_F, HID_F);
    fill_kernel<<<cdiv(N_NODES, 256), b256, 0, stream>>>(SMAX, ninf, N_NODES);
    fill_kernel<<<cdiv(N_NODES, 256), b256, 0, stream>>>(DEN, 0.0f, N_NODES);
    fill_kernel<<<cdiv((long long)N_NODES * HID_F, 256), b256, 0, stream>>>(AGG, 0.0f, (long long)N_NODES * HID_F);
    edge_scores_kernel<<<waveBlocks, b256, 0, stream>>>(ei, EA, LOOPA, XL, XR, we2, att2, S, SMAX, etot, HID_F);
    edge_softmax_kernel<<<cdiv(etot, 256), b256, 0, stream>>>(ei, EA, LOOPA, S, SMAX, ES, DEN, etot);
    edge_aggregate_kernel<<<waveBlocks, b256, 0, stream>>>(ei, EA, LOOPA, XL, ES, DEN, AGG, etot, HID_F);

    // ===== layer 3: K=256, C=128 =====
    prep_kernel<<<cdiv((long long)N_NODES * HID_F, 256), b256, 0, stream>>>(
        AGG, b2, XBF, (long long)N_NODES * HID_F, HID_F, 0x85EBCA77u);
    conv_bf16_kernel<<<cdiv(HID_F * OUT_F, 256), b256, 0, stream>>>(wl3, WLB, HID_F * OUT_F);
    conv_bf16_kernel<<<cdiv(HID_F * OUT_F, 256), b256, 0, stream>>>(wr3, WRB, HID_F * OUT_F);
    wmma_dual_gemm<<<dim3(cdiv(N_NODES, 128), OUT_F / 64), b256, 0, stream>>>(
        XBF, WLB, WRB, XL, XR, N_NODES, HID_F, OUT_F);
    fill_kernel<<<cdiv(N_NODES, 256), b256, 0, stream>>>(SMAX, ninf, N_NODES);
    fill_kernel<<<cdiv(N_NODES, 256), b256, 0, stream>>>(DEN, 0.0f, N_NODES);
    fill_kernel<<<cdiv((long long)N_NODES * OUT_F, 256), b256, 0, stream>>>(AGG, 0.0f, (long long)N_NODES * OUT_F);
    edge_scores_kernel<<<waveBlocks, b256, 0, stream>>>(ei, EA, LOOPA, XL, XR, we3, att3, S, SMAX, etot, OUT_F);
    edge_softmax_kernel<<<cdiv(etot, 256), b256, 0, stream>>>(ei, EA, LOOPA, S, SMAX, ES, DEN, etot);
    edge_aggregate_kernel<<<waveBlocks, b256, 0, stream>>>(ei, EA, LOOPA, XL, ES, DEN, AGG, etot, OUT_F);

    final_bias_kernel<<<cdiv((long long)N_NODES * OUT_F, 256), b256, 0, stream>>>(
        AGG, b3, (float*)d_out, (long long)N_NODES * OUT_F, OUT_F);
}
